// VectorQuantizer_19610820673825
// MI455X (gfx1250) — compile-verified, matrix operand order checked by reading, not run on hardware
//
#include <hip/hip_runtime.h>
#include <hip/hip_bf16.h>
#include <math.h>

// ---------- problem constants ----------
#define BATCH   16
#define DIM     256
#define TLEN    2048
#define KCODES  2048
#define NROWS   (BATCH * TLEN)        // 32768
#define QELEMS  (BATCH * DIM * TLEN)  // 8388608
#define LOSS_OFF ((size_t)QELEMS)
#define IDX_OFF  ((size_t)QELEMS + 1)
#define PPL_OFF  ((size_t)QELEMS + 1 + NROWS)

#define MBLK 32
#define NWG  (NROWS / MBLK)           // 1024

// ---------- workspace layout (bytes) ----------
#define WS_EMB_HI 0
#define WS_EMB_LO ((size_t)KCODES * DIM * 2)                 // 1 MB
#define WS_E2     (WS_EMB_LO + (size_t)KCODES * DIM * 2)     // 2 MB
#define WS_CNT    (WS_E2 + (size_t)KCODES * 4)
#define WS_PART   (WS_CNT + (size_t)KCODES * 4)

typedef __attribute__((ext_vector_type(16))) __bf16 v16bf;
typedef __attribute__((ext_vector_type(8)))  float  v8f;

union V16 { v16bf v; unsigned u[8]; unsigned short s[16]; };

__device__ __forceinline__ unsigned short f32_to_bf16_rne(float f) {
    unsigned u = __float_as_uint(f);
    unsigned r = u + 0x7fffu + ((u >> 16) & 1u);   // round to nearest even
    return (unsigned short)(r >> 16);
}
__device__ __forceinline__ float bf16_bits_to_f32(unsigned short h) {
    return __uint_as_float(((unsigned)h) << 16);
}
// monotonic (totally ordered) uint key for float
__device__ __forceinline__ unsigned fkey(float f) {
    unsigned u = __float_as_uint(f);
    return (u & 0x80000000u) ? ~u : (u | 0x80000000u);
}
__device__ __forceinline__ unsigned long long shfl_xor_u64(unsigned long long v, int mask) {
    unsigned lo = (unsigned)v, hi = (unsigned)(v >> 32);
    lo = __shfl_xor(lo, mask, 32);
    hi = __shfl_xor(hi, mask, 32);
    return ((unsigned long long)hi << 32) | lo;
}

// ---------------- kernel 0: codebook prep ----------------
// block k (K blocks x 256 threads): split e -> bf16 hi/lo, ||e||^2, zero counts
__global__ __launch_bounds__(256) void vq_prep(const float* __restrict__ emb,
                                               unsigned short* __restrict__ ehi,
                                               unsigned short* __restrict__ elo,
                                               float* __restrict__ e2,
                                               unsigned* __restrict__ counts) {
    __shared__ float red[8];
    const int k = blockIdx.x;
    const int d = threadIdx.x;
    const size_t off = (size_t)k * DIM + d;
    float e = emb[off];
    unsigned short hb = f32_to_bf16_rne(e);
    ehi[off] = hb;
    elo[off] = f32_to_bf16_rne(e - bf16_bits_to_f32(hb));
    float sq = e * e;
    #pragma unroll
    for (int m = 16; m >= 1; m >>= 1) sq += __shfl_xor(sq, m, 32);
    if ((d & 31) == 0) red[d >> 5] = sq;
    __syncthreads();
    if (d == 0) {
        float s = 0.f;
        #pragma unroll
        for (int i = 0; i < 8; ++i) s += red[i];
        e2[k] = s;
        counts[k] = 0u;
    }
}

// ---------------- kernel 1: distances + argmin + quantize ----------------
__global__ __launch_bounds__(256, 1) void vq_main(const float* __restrict__ x,
                                                  const float* __restrict__ emb,
                                                  float* __restrict__ out,
                                                  const unsigned short* __restrict__ ehi,
                                                  const unsigned short* __restrict__ elo,
                                                  const float* __restrict__ e2,
                                                  unsigned* __restrict__ counts,
                                                  float* __restrict__ partials) {
    __shared__ unsigned long long best[MBLK];
    __shared__ unsigned idxs[MBLK];
    __shared__ float wsum[8];

    const int tid   = threadIdx.x;
    const int lane  = tid & 31;
    const int wave  = tid >> 5;
    const int half  = lane >> 4;     // K-half selector for A/B operand layout
    const int nl    = lane & 15;     // A: row m ; B/C: column n
    const int mtile = wave >> 2;     // 0..1  (row tile of 16)
    const int ktile = wave & 3;      // 0..3  (col tile of 16 inside 64-chunk)
    const int wg    = blockIdx.x;
    const int n0    = wg * MBLK;
    const int bb    = n0 >> 11;      // / TLEN  (MBLK divides TLEN -> single b)
    const int t0    = n0 & (TLEN - 1);

    if (tid < MBLK) best[tid] = ~0ULL;
    __syncthreads();

    // ---- hoist A fragments (x rows, bf16 hi/lo) for full D into registers ----
    V16 ahi[8], alo[8];
    {
        const int trow = t0 + mtile * 16 + nl;                  // lane's x-row (t index)
        const float* xb = x + (size_t)bb * DIM * TLEN + trow;   // + d*TLEN indexing
        #pragma unroll
        for (int dsb = 0; dsb < 8; ++dsb) {
            #pragma unroll
            for (int e = 0; e < 16; ++e) {
                // 16-bit A 16x32 layout: elem e -> K = (e&7) + (e>=8 ? 16:0) + 8*half
                int dl = (e & 7) + ((e >> 3) << 4) + (half << 3);
                float xv = xb[(size_t)(dsb * 32 + dl) * TLEN];
                unsigned short hb = f32_to_bf16_rne(xv);
                ahi[dsb].s[e] = hb;
                alo[dsb].s[e] = f32_to_bf16_rne(xv - bf16_bits_to_f32(hb));
            }
        }
    }

    unsigned long long run[8];
    #pragma unroll
    for (int r = 0; r < 8; ++r) run[r] = ~0ULL;

    // ---- sweep codebook in chunks of 64 (each wave owns 16 columns) ----
    for (int chunk = 0; chunk < KCODES / 64; ++chunk) {
        const int k0 = chunk * 64 + ktile * 16;
        const unsigned* bhrow = (const unsigned*)ehi + (((size_t)(k0 + nl) * DIM) >> 1);
        const unsigned* blrow = (const unsigned*)elo + (((size_t)(k0 + nl) * DIM) >> 1);

        v8f acc0 = {0.f,0.f,0.f,0.f,0.f,0.f,0.f,0.f};
        v8f acc1 = {0.f,0.f,0.f,0.f,0.f,0.f,0.f,0.f};
        v8f acc2 = {0.f,0.f,0.f,0.f,0.f,0.f,0.f,0.f};

        #pragma unroll
        for (int dsb = 0; dsb < 8; ++dsb) {
            V16 bhi, blo;
            #pragma unroll
            for (int i = 0; i < 8; ++i) {
                // B 32x16 layout (K striped like A): dword i -> K pair at:
                int dl = 2 * i + (half << 3) + ((i >= 4) ? 8 : 0);
                bhi.u[i] = bhrow[(dsb * 32 + dl) >> 1];
                blo.u[i] = blrow[(dsb * 32 + dl) >> 1];
            }
            // three independent accumulation chains: hi*hi + lo*hi + hi*lo
            acc0 = __builtin_amdgcn_wmma_f32_16x16x32_bf16(false, ahi[dsb].v, false, bhi.v,
                                                           (short)0, acc0, false, false);
            acc1 = __builtin_amdgcn_wmma_f32_16x16x32_bf16(false, alo[dsb].v, false, bhi.v,
                                                           (short)0, acc1, false, false);
            acc2 = __builtin_amdgcn_wmma_f32_16x16x32_bf16(false, ahi[dsb].v, false, blo.v,
                                                           (short)0, acc2, false, false);
        }

        const float e2v = e2[k0 + nl];
        const unsigned ncode = (unsigned)(k0 + nl);
        #pragma unroll
        for (int r = 0; r < 8; ++r) {
            float dot = acc0[r] + acc1[r] + acc2[r];
            float d2 = fmaf(-2.0f, dot, e2v);          // ||x||^2 dropped (row-constant)
            unsigned long long key = ((unsigned long long)fkey(d2) << 32) | ncode;
            run[r] = run[r] < key ? run[r] : key;
        }
    }

    // ---- reduce argmin: lanes within each 16-group, then ds_min_u64 ----
    #pragma unroll
    for (int r = 0; r < 8; ++r) {
        unsigned long long v = run[r];
        #pragma unroll
        for (int m = 8; m >= 1; m >>= 1) {
            unsigned long long o = shfl_xor_u64(v, m);
            v = v < o ? v : o;
        }
        if (nl == 0) atomicMin(&best[mtile * 16 + r + (half << 3)], v);
    }
    __syncthreads();

    if (tid < MBLK) {
        unsigned id = (unsigned)(best[tid] & 0xffffffffu);
        idxs[tid] = id;
        out[IDX_OFF + (size_t)n0 + tid] = (float)id;
        atomicAdd(&counts[id], 1u);
    }
    __syncthreads();

    // ---- quantized output (exact f32 gather) + squared-error partial ----
    float sq = 0.f;
    {
        const int tl = tid & 31;              // row within block (t offset)
        const int dr = tid >> 5;              // 0..7
        const unsigned id = idxs[tl];
        const float* erow = emb + (size_t)id * DIM;
        const size_t base = (size_t)bb * DIM * TLEN + (size_t)(t0 + tl);
        #pragma unroll 4
        for (int j = 0; j < 32; ++j) {
            int d = dr + j * 8;
            float q = erow[d];
            size_t off = base + (size_t)d * TLEN;
            float diff = q - x[off];
            out[off] = q;
            sq = fmaf(diff, diff, sq);
        }
    }
    #pragma unroll
    for (int m = 16; m >= 1; m >>= 1) sq += __shfl_xor(sq, m, 32);
    if (lane == 0) wsum[wave] = sq;
    __syncthreads();
    if (tid == 0) {
        float s = 0.f;
        #pragma unroll
        for (int i = 0; i < 8; ++i) s += wsum[i];   // deterministic order
        partials[wg] = s;
    }
}

// ---------------- kernel 2: loss + perplexity ----------------
__global__ __launch_bounds__(256) void vq_final(const float* __restrict__ partials,
                                                const unsigned* __restrict__ counts,
                                                float* __restrict__ out) {
    __shared__ float r1[8], r2[8];
    const int tid = threadIdx.x;
    float ls = 0.f;
    for (int i = tid; i < NWG; i += 256) ls += partials[i];
    float ps = 0.f;
    for (int k = tid; k < KCODES; k += 256) {
        float p = (float)counts[k] * (1.0f / (float)NROWS);
        ps += p * logf(p + 1e-10f);
    }
    #pragma unroll
    for (int m = 16; m >= 1; m >>= 1) {
        ls += __shfl_xor(ls, m, 32);
        ps += __shfl_xor(ps, m, 32);
    }
    if ((tid & 31) == 0) { r1[tid >> 5] = ls; r2[tid >> 5] = ps; }
    __syncthreads();
    if (tid == 0) {
        float tls = 0.f, tps = 0.f;
        #pragma unroll
        for (int i = 0; i < 8; ++i) { tls += r1[i]; tps += r2[i]; }
        out[LOSS_OFF] = 1.25f * tls / (float)QELEMS;  // q_loss + 0.25*e_loss
        out[PPL_OFF]  = expf(-tps);
    }
}

// ---------------- host launch ----------------
extern "C" void kernel_launch(void* const* d_in, const int* in_sizes, int n_in,
                              void* d_out, int out_size, void* d_ws, size_t ws_size,
                              hipStream_t stream) {
    (void)in_sizes; (void)n_in; (void)out_size; (void)ws_size;
    const float* x   = (const float*)d_in[0];   // [16, 256, 2048] f32
    const float* emb = (const float*)d_in[1];   // [2048, 256] f32
    float* out = (float*)d_out;

    char* ws = (char*)d_ws;
    unsigned short* ehi = (unsigned short*)(ws + WS_EMB_HI);
    unsigned short* elo = (unsigned short*)(ws + WS_EMB_LO);
    float*    e2       = (float*)(ws + WS_E2);
    unsigned* counts   = (unsigned*)(ws + WS_CNT);
    float*    partials = (float*)(ws + WS_PART);

    vq_prep<<<dim3(KCODES), dim3(256), 0, stream>>>(emb, ehi, elo, e2, counts);
    vq_main<<<dim3(NWG), dim3(256), 0, stream>>>(x, emb, out, ehi, elo, e2, counts, partials);
    vq_final<<<dim3(1), dim3(256), 0, stream>>>(partials, counts, out);
}
